// PointNet2PTMSGDynamic_38268158607591
// MI455X (gfx1250) — matrix operand order repaired and checked by reading, not compiled, not run on hardware
//
#include <hip/hip_runtime.h>

typedef __attribute__((ext_vector_type(16))) _Float16 v16h;
typedef __attribute__((ext_vector_type(8)))  float    v8f;

union H16 { v16h v; _Float16 s[16]; };

#define NPIECE 16

// ---------------------------------------------------------------------------
// utility
// ---------------------------------------------------------------------------
__global__ __launch_bounds__(256) void zero_kernel(float* __restrict__ p, long n) {
  long i = (long)blockIdx.x * 256 + threadIdx.x;
  if (i < n) p[i] = 0.0f;
}

// ---------------------------------------------------------------------------
// farthest point sampling: one block per piece
// ---------------------------------------------------------------------------
__global__ __launch_bounds__(256)
void fps_kernel(const float* __restrict__ xyz, int n, int S,
                int* __restrict__ oidx, float* __restrict__ oxyz) {
  __shared__ float dist[4096];
  __shared__ float rv[256];
  __shared__ int   ri[256];
  const int b = blockIdx.x, tid = threadIdx.x;
  const float* P = xyz + (long)b * n * 3;
  for (int i = tid; i < n; i += 256) dist[i] = 3.0e38f;
  if (tid == 0) {
    oidx[(long)b * S] = 0;
    oxyz[((long)b * S) * 3 + 0] = P[0];
    oxyz[((long)b * S) * 3 + 1] = P[1];
    oxyz[((long)b * S) * 3 + 2] = P[2];
  }
  __syncthreads();
  int last = 0;
  for (int t = 1; t < S; ++t) {
    const float lx = P[last * 3 + 0], ly = P[last * 3 + 1], lz = P[last * 3 + 2];
    float bv = -1.0f; int bi = n;
    for (int i = tid; i < n; i += 256) {
      float dx = P[i * 3 + 0] - lx, dy = P[i * 3 + 1] - ly, dz = P[i * 3 + 2] - lz;
      float d = dx * dx + dy * dy + dz * dz;
      float dd = fminf(dist[i], d);
      dist[i] = dd;
      if (dd > bv) { bv = dd; bi = i; }
    }
    rv[tid] = bv; ri[tid] = bi;
    __syncthreads();
    for (int s2 = 128; s2 > 0; s2 >>= 1) {
      if (tid < s2) {
        float ov = rv[tid + s2]; int oi = ri[tid + s2];
        if (ov > rv[tid] || (ov == rv[tid] && oi < ri[tid])) { rv[tid] = ov; ri[tid] = oi; }
      }
      __syncthreads();
    }
    last = ri[0];
    if (tid == 0) {
      oidx[(long)b * S + t] = last;
      oxyz[((long)b * S + t) * 3 + 0] = P[last * 3 + 0];
      oxyz[((long)b * S + t) * 3 + 1] = P[last * 3 + 1];
      oxyz[((long)b * S + t) * 3 + 2] = P[last * 3 + 2];
    }
    __syncthreads();
  }
}

// ---------------------------------------------------------------------------
// 32 nearest neighbors (sorted ascending) per center: one wave per center
// ---------------------------------------------------------------------------
__global__ __launch_bounds__(32)
void knn32_kernel(const float* __restrict__ xyz, const float* __restrict__ cent,
                  int n, int S, int* __restrict__ nidx, float* __restrict__ nd) {
  __shared__ float d2s[4096];
  const int c = blockIdx.x;           // center id in [0, NPIECE*S)
  const int lane = threadIdx.x;
  const int b = c / S;
  const float* P = xyz + (long)b * n * 3;
  const float cx = cent[(long)c * 3 + 0], cy = cent[(long)c * 3 + 1], cz = cent[(long)c * 3 + 2];
  for (int i = lane; i < n; i += 32) {
    float dx = P[i * 3 + 0] - cx, dy = P[i * 3 + 1] - cy, dz = P[i * 3 + 2] - cz;
    d2s[i] = dx * dx + dy * dy + dz * dz;
  }
  __syncthreads();
  for (int t = 0; t < 32; ++t) {
    float bd = 3.0e38f; int bi = n;
    for (int i = lane; i < n; i += 32) {
      float v = d2s[i];
      if (v < bd) { bd = v; bi = i; }
    }
    for (int off = 16; off > 0; off >>= 1) {
      float od = __shfl_xor(bd, off);
      int   oi = __shfl_xor(bi, off);
      if (od < bd || (od == bd && oi < bi)) { bd = od; bi = oi; }
    }
    if (lane == 0) {
      nidx[(long)c * 32 + t] = bi;
      nd[(long)c * 32 + t] = bd;
      d2s[bi] = 3.0e38f;
    }
    __syncthreads();
  }
}

// ---------------------------------------------------------------------------
// gather grouped (centered-xyz ++ feats) into WMMA A-fragment layout (f16)
// A layout: [mtile][ktile][lane(32)][e(16)]; K(lane,e)=((e&8)<<1)+((lane>>4)<<3)+(e&7)
// ---------------------------------------------------------------------------
__global__ __launch_bounds__(256)
void pack_grouped_kernel(const float* __restrict__ xyz, const float* __restrict__ feats,
                         const float* __restrict__ cent,
                         const int* __restrict__ nidx, const float* __restrict__ nd,
                         float r2, int k, int S, int n, int C, int Kt, int Mtiles,
                         _Float16* __restrict__ A) {
  int tid = blockIdx.x * 256 + threadIdx.x;
  if (tid >= Mtiles * 32) return;
  const int mt = tid >> 5, lane = tid & 31;
  const int M = lane & 15, hi = (lane >> 4) << 3;
  const long row = (long)mt * 16 + M;
  const long center = row / k;
  const int  j = (int)(row % k);
  const int  b = (int)(center / S);
  const int* nx = nidx + center * 32;
  int pidx = nx[j];
  if (nd[center * 32 + j] > r2) pidx = nx[0];     // ball query: out-of-radius -> nearest
  const float* ct = cent + center * 3;
  const float* P = xyz + ((long)b * n + pidx) * 3;
  float g0 = P[0] - ct[0], g1 = P[1] - ct[1], g2 = P[2] - ct[2];
  const float* F = feats + ((long)b * n + pidx) * C;
  const int cinp = C + 3;
  for (int kt = 0; kt < Kt; ++kt) {
    H16 pk;
#pragma unroll
    for (int e = 0; e < 16; ++e) {
      int Kg = kt * 32 + ((e & 8) << 1) + hi + (e & 7);
      float v = 0.0f;
      if (Kg == 0) v = g0;
      else if (Kg == 1) v = g1;
      else if (Kg == 2) v = g2;
      else if (Kg < cinp) v = F[Kg - 3];
      pk.s[e] = (_Float16)v;
    }
    *(v16h*)(A + ((long)mt * Kt + kt) * 512 + lane * 16) = pk.v;
  }
}

// row-major f32 -> A-fragment layout
__global__ __launch_bounds__(256)
void pack_rows_kernel(const float* __restrict__ src, long rows, int K, int Kt, int Mtiles,
                      _Float16* __restrict__ A) {
  int tid = blockIdx.x * 256 + threadIdx.x;
  if (tid >= Mtiles * 32) return;
  const int mt = tid >> 5, lane = tid & 31;
  const int M = lane & 15, hi = (lane >> 4) << 3;
  const long row = (long)mt * 16 + M;
  for (int kt = 0; kt < Kt; ++kt) {
    H16 pk;
#pragma unroll
    for (int e = 0; e < 16; ++e) {
      int Kg = kt * 32 + ((e & 8) << 1) + hi + (e & 7);
      float v = (row < rows && Kg < K) ? src[row * K + Kg] : 0.0f;
      pk.s[e] = (_Float16)v;
    }
    *(v16h*)(A + ((long)mt * Kt + kt) * 512 + lane * 16) = pk.v;
  }
}

// W[N,K] f32 -> B-fragment layout: [ntile16][ktile][lane(32), N=lane&15][e(16) along K, same K order as A]
__global__ __launch_bounds__(256)
void pack_w_kernel(const float* __restrict__ W, int N, int K, int Ntiles, int Kt,
                   _Float16* __restrict__ Wp) {
  int tid = blockIdx.x * 256 + threadIdx.x;
  if (tid >= Ntiles * 32) return;
  const int nt = tid >> 5, lane = tid & 31;
  for (int kt = 0; kt < Kt; ++kt) {
    H16 pk;
#pragma unroll
    for (int e = 0; e < 16; ++e) {
      int nn = nt * 16 + (lane & 15), kkk = kt * 32 + ((e & 8) << 1) + ((lane >> 4) << 3) + (e & 7);
      pk.s[e] = (_Float16)((nn < N && kkk < K) ? W[(long)nn * K + kkk] : 0.0f);
    }
    *(v16h*)(Wp + ((long)nt * Kt + kt) * 512 + lane * 16) = pk.v;
  }
}

// ---------------------------------------------------------------------------
// WMMA GEMM: D[M,N] = relu(A[M,K] * W^T + bias).  One wave per (mtile, 32-N chunk).
// Epilogue bounces through LDS; outputs packed-A (next layer), row-major f32,
// and/or fused group max-pool via integer atomic max (valid: relu >= 0, buf zeroed).
// ---------------------------------------------------------------------------
__global__ __launch_bounds__(256)
void wmma_gemm_kernel(const _Float16* __restrict__ A, const _Float16* __restrict__ Wp,
                      const float* __restrict__ bias,
                      int Nvalid, int Mtiles, int Nchunks, int Kt,
                      _Float16* __restrict__ Aout, float* __restrict__ rowout,
                      float* __restrict__ pool, int poolK, int poolCtot, int poolCoff,
                      int relu) {
  __shared__ float lds[8][512];
  const int wave = threadIdx.x >> 5;
  const int lane = threadIdx.x & 31;
  const int unit = blockIdx.x * 8 + wave;
  if (unit >= Mtiles * Nchunks) return;
  const int mt = unit / Nchunks;
  const int nc = unit % Nchunks;

  v8f acc0 = {0.f,0.f,0.f,0.f,0.f,0.f,0.f,0.f};
  v8f acc1 = {0.f,0.f,0.f,0.f,0.f,0.f,0.f,0.f};
  const _Float16* ap  = A  + ((long)mt * Kt) * 512 + lane * 16;
  const _Float16* b0p = Wp + ((long)(nc * 2 + 0) * Kt) * 512 + lane * 16;
  const _Float16* b1p = Wp + ((long)(nc * 2 + 1) * Kt) * 512 + lane * 16;
  for (int kt = 0; kt < Kt; ++kt) {
    v16h a  = *(const v16h*)(ap  + (long)kt * 512);
    v16h b0 = *(const v16h*)(b0p + (long)kt * 512);
    v16h b1 = *(const v16h*)(b1p + (long)kt * 512);
    acc0 = __builtin_amdgcn_wmma_f32_16x16x32_f16(false, a, false, b0, (short)0, acc0, false, false);
    acc1 = __builtin_amdgcn_wmma_f32_16x16x32_f16(false, a, false, b1, (short)0, acc1, false, false);
  }

  const int ncol = lane & 15;
  const int mhi  = (lane >> 4) << 3;
  float bv0 = 0.f, bv1 = 0.f;
  {
    int n0 = nc * 32 + ncol, n1 = nc * 32 + 16 + ncol;
    if (n0 < Nvalid) bv0 = bias[n0];
    if (n1 < Nvalid) bv1 = bias[n1];
  }
  float* L = lds[wave];
#pragma unroll
  for (int r = 0; r < 8; ++r) {           // D layout: M = r + 8*(lane>=16), N = lane&15
    float h0 = acc0[r] + bv0;
    float h1 = acc1[r] + bv1;
    if (relu) { h0 = fmaxf(h0, 0.f); h1 = fmaxf(h1, 0.f); }
    int M = r + mhi;
    L[M * 32 + ncol]      = h0;
    L[M * 32 + 16 + ncol] = h1;
  }
  asm volatile("s_wait_dscnt 0" ::: "memory");   // same-wave LDS RAW fence

  if (Aout) {                              // repack 16x32 tile as next-layer A fragment
    H16 pk;
    const int M = lane & 15;
#pragma unroll
    for (int e = 0; e < 16; ++e) {
      int K = ((e & 8) << 1) + mhi + (e & 7);
      pk.s[e] = (_Float16)L[M * 32 + K];
    }
    *(v16h*)(Aout + ((long)mt * Nchunks + nc) * 512 + lane * 16) = pk.v;
  }
  if (rowout) {
    int col = nc * 32 + lane;
    if (col < Nvalid) {
#pragma unroll 4
      for (int i = 0; i < 16; ++i) {
        long row = (long)mt * 16 + i;
        rowout[row * Nvalid + col] = L[i * 32 + lane];
      }
    }
  }
  if (pool) {
    int col = nc * 32 + lane;
    if (col < Nvalid) {
      for (int i = 0; i < 16; ++i) {
        long row = (long)mt * 16 + i;
        long center = row / poolK;
        atomicMax((int*)(pool + center * poolCtot + poolCoff + col),
                  __float_as_int(L[i * 32 + lane]));
      }
    }
  }
}

// ---------------------------------------------------------------------------
// FP: 3-NN + inverse-distance weights (thread per fine point)
// ---------------------------------------------------------------------------
__global__ __launch_bounds__(256)
void knn3_kernel(const float* __restrict__ xf, const float* __restrict__ xc,
                 int n1, int n2, int* __restrict__ idx3, float* __restrict__ w3,
                 long rows) {
  long row = (long)blockIdx.x * 256 + threadIdx.x;
  if (row >= rows) return;
  const int b = (int)(row / n1);
  const float* p = xf + row * 3;
  const float* Q = xc + (long)b * n2 * 3;
  float px = p[0], py = p[1], pz = p[2];
  float d0 = 3.0e38f, d1 = 3.0e38f, d2v = 3.0e38f;
  int i0 = 0, i1 = 0, i2 = 0;
  for (int m = 0; m < n2; ++m) {
    float dx = Q[m * 3 + 0] - px, dy = Q[m * 3 + 1] - py, dz = Q[m * 3 + 2] - pz;
    float d = dx * dx + dy * dy + dz * dz;
    if (d < d0)      { d2v = d1; i2 = i1; d1 = d0; i1 = i0; d0 = d; i0 = m; }
    else if (d < d1) { d2v = d1; i2 = i1; d1 = d; i1 = m; }
    else if (d < d2v){ d2v = d; i2 = m; }
  }
  float w0 = 1.0f / (fmaxf(d0, 0.f) + 1e-8f);
  float w1 = 1.0f / (fmaxf(d1, 0.f) + 1e-8f);
  float w2 = 1.0f / (fmaxf(d2v, 0.f) + 1e-8f);
  float wsum = w0 + w1 + w2;
  idx3[row * 3 + 0] = i0; idx3[row * 3 + 1] = i1; idx3[row * 3 + 2] = i2;
  w3[row * 3 + 0] = w0 / wsum; w3[row * 3 + 1] = w1 / wsum; w3[row * 3 + 2] = w2 / wsum;
}

__global__ __launch_bounds__(256)
void interp_concat_kernel(const float* __restrict__ f1, int C1,
                          const float* __restrict__ f2, int C2,
                          const int* __restrict__ idx3, const float* __restrict__ w3,
                          long rows, int n1, int n2, float* __restrict__ out) {
  const int Kc = C1 + C2;
  long gid = (long)blockIdx.x * 256 + threadIdx.x;
  if (gid >= rows * Kc) return;
  long row = gid / Kc;
  int c = (int)(gid % Kc);
  float v;
  if (c < C1) {
    v = f1[row * C1 + c];
  } else {
    const int b = (int)(row / n1);
    const int c2 = c - C1;
    const float* F2 = f2 + (long)b * n2 * C2;
    const int* ix = idx3 + row * 3;
    const float* ww = w3 + row * 3;
    v = ww[0] * F2[(long)ix[0] * C2 + c2]
      + ww[1] * F2[(long)ix[1] * C2 + c2]
      + ww[2] * F2[(long)ix[2] * C2 + c2];
  }
  out[row * Kc + c] = v;
}

__global__ __launch_bounds__(256)
void head_kernel(const float* __restrict__ f, const float* __restrict__ W,
                 const float* __restrict__ bb, float* __restrict__ out, long rows) {
  long r = (long)blockIdx.x * 256 + threadIdx.x;
  if (r >= rows) return;
  const float* xr = f + r * 128;
  float a0 = bb[0], a1 = bb[1];
  for (int i = 0; i < 128; ++i) {
    float xi = xr[i];
    a0 += W[i] * xi;
    a1 += W[128 + i] * xi;
  }
  out[r * 2 + 0] = a0;
  out[r * 2 + 1] = a1;
}

// ---------------------------------------------------------------------------
// host orchestration
// ---------------------------------------------------------------------------
static void run_mlp(hipStream_t stream, const float* const* Ws, const float* const* Bs,
                    const int* dims, int nl, int cin, int rows,
                    _Float16* bufA, _Float16* bufB, _Float16* Wp,
                    float* rowout, float* pool, int poolK, int poolCtot, int poolCoff) {
  _Float16* cur = bufA;
  _Float16* nxt = bufB;
  const int Mtiles = rows / 16;
  int K = cin;
  for (int l = 0; l < nl; ++l) {
    const int N = dims[l];
    const int Kpad = (K + 31) & ~31;
    const int Kt = Kpad / 32;
    const int Npad = (N + 31) & ~31;
    const int Nchunks = Npad / 32;
    const int Ntiles = Npad / 16;
    pack_w_kernel<<<(Ntiles * 32 + 255) / 256, 256, 0, stream>>>(Ws[l], N, K, Ntiles, Kt, Wp);
    const bool last = (l == nl - 1);
    const int units = Mtiles * Nchunks;
    wmma_gemm_kernel<<<(units + 7) / 8, 256, 0, stream>>>(
        cur, Wp, Bs[l], N, Mtiles, Nchunks, Kt,
        last ? nullptr : nxt,
        last ? rowout : nullptr,
        last ? pool : nullptr, poolK, poolCtot, poolCoff, 1);
    _Float16* t = cur; cur = nxt; nxt = t;
    K = N;
  }
}

extern "C" void kernel_launch(void* const* d_in, const int* in_sizes, int n_in,
                              void* d_out, int out_size, void* d_ws, size_t ws_size,
                              hipStream_t stream) {
  (void)in_sizes; (void)n_in; (void)out_size; (void)ws_size;
  static const int Sl[5] = {4096, 615, 154, 39, 10};      // points per level
  static const int Cl[5] = {3, 96, 256, 512, 1024};       // feature channels per level
  static const float R2[4][2] = {{0.0025f, 0.01f}, {0.01f, 0.04f}, {0.04f, 0.16f}, {0.16f, 0.64f}};
  static const int KK[2] = {16, 32};
  static const int saDims[4][2][3] = {
      {{16, 16, 32}, {32, 32, 64}},
      {{64, 64, 128}, {64, 96, 128}},
      {{128, 196, 256}, {128, 196, 256}},
      {{256, 256, 512}, {256, 384, 512}}};
  static const int fpDims[4][3] = {{256, 256, 0}, {256, 256, 0}, {256, 128, 0}, {128, 128, 128}};
  static const int fpNL[4] = {2, 2, 2, 3};

  const float* x = (const float*)d_in[0];

  int pi = 1;
  const float *saW[4][2][3], *saB[4][2][3];
  for (int l = 0; l < 4; ++l)
    for (int s = 0; s < 2; ++s)
      for (int m = 0; m < 3; ++m) {
        saW[l][s][m] = (const float*)d_in[pi++];
        saB[l][s][m] = (const float*)d_in[pi++];
      }
  const float *fpW[4][3] = {}, *fpB[4][3] = {};
  for (int l = 0; l < 4; ++l)
    for (int m = 0; m < fpNL[l]; ++m) {
      fpW[l][m] = (const float*)d_in[pi++];
      fpB[l][m] = (const float*)d_in[pi++];
    }
  const float* headW = (const float*)d_in[pi++];
  const float* headB = (const float*)d_in[pi++];

  // scratch carve-out
  char* base = (char*)d_ws;
  size_t off = 0;
  auto alloc = [&](size_t bytes) -> void* {
    void* p = base + off;
    off = (off + bytes + 255) & ~(size_t)255;
    return p;
  };
  const float* XYZ[5];
  float* xyzbuf[5];
  float* FEAT[5];
  XYZ[0] = x; xyzbuf[0] = nullptr; FEAT[0] = nullptr;
  for (int l = 1; l < 5; ++l) {
    xyzbuf[l] = (float*)alloc((size_t)NPIECE * Sl[l] * 3 * 4);
    XYZ[l] = xyzbuf[l];
    FEAT[l] = (float*)alloc((size_t)NPIECE * Sl[l] * Cl[l] * 4);
  }
  float* FN3 = (float*)alloc((size_t)NPIECE * 39 * 256 * 4);
  float* FN2 = (float*)alloc((size_t)NPIECE * 154 * 256 * 4);
  float* FN1 = (float*)alloc((size_t)NPIECE * 615 * 128 * 4);
  float* FN0 = (float*)alloc((size_t)NPIECE * 4096 * 128 * 4);
  int*   fpsidx = (int*)alloc((size_t)NPIECE * 615 * 4);
  int*   nidxb  = (int*)alloc((size_t)NPIECE * 615 * 32 * 4);
  float* ndb    = (float*)alloc((size_t)NPIECE * 615 * 32 * 4);
  int*   idx3   = (int*)alloc((size_t)NPIECE * 4096 * 3 * 4);
  float* w3     = (float*)alloc((size_t)NPIECE * 4096 * 3 * 4);
  _Float16* Wp  = (_Float16*)alloc(1u << 20);          // max packed W: 256x1536 f16
  _Float16* bufA = (_Float16*)alloc(25u << 20);        // max packed A ~20.2 MB
  _Float16* bufB = (_Float16*)alloc(35u << 20);        // packed ping-pong; aliases hcat

  // ------------------------- SA stages -------------------------
  for (int l = 0; l < 4; ++l) {
    const int n = Sl[l], S = Sl[l + 1], C = Cl[l];
    const float* pxyz = XYZ[l];
    const float* pfeat = (l == 0) ? x : FEAT[l];
    float* oxyz = xyzbuf[l + 1];
    float* ofeat = FEAT[l + 1];
    fps_kernel<<<NPIECE, 256, 0, stream>>>(pxyz, n, S, fpsidx, oxyz);
    knn32_kernel<<<NPIECE * S, 32, 0, stream>>>(pxyz, oxyz, n, S, nidxb, ndb);
    const long fz = (long)NPIECE * S * Cl[l + 1];
    zero_kernel<<<(int)((fz + 255) / 256), 256, 0, stream>>>(ofeat, fz);
    int coff = 0;
    for (int sc = 0; sc < 2; ++sc) {
      const int k = KK[sc];
      const int rows = NPIECE * S * k;
      const int cinp = C + 3;
      const int Kpad = (cinp + 31) & ~31;
      const int Kt = Kpad / 32;
      const int Mtiles = rows / 16;
      pack_grouped_kernel<<<(Mtiles * 32 + 255) / 256, 256, 0, stream>>>(
          pxyz, pfeat, oxyz, nidxb, ndb, R2[l][sc], k, S, n, C, Kt, Mtiles, bufA);
      run_mlp(stream, saW[l][sc], saB[l][sc], saDims[l][sc], 3, cinp, rows,
              bufA, bufB, Wp, nullptr, ofeat, k, Cl[l + 1], coff);
      coff += saDims[l][sc][2];
    }
  }

  // ------------------------- FP stages -------------------------
  struct FpSpec { int fl, cl; const float* f1; int C1; const float* f2; int C2; float* out; };
  FpSpec spec[4] = {
      {3, 4, FEAT[3], 512, FEAT[4], 1024, FN3},
      {2, 3, FEAT[2], 256, FN3, 256, FN2},
      {1, 2, FEAT[1], 96, FN2, 256, FN1},
      {0, 1, nullptr, 0, FN1, 128, FN0}};
  for (int i = 0; i < 4; ++i) {
    const int n1 = Sl[spec[i].fl], n2 = Sl[spec[i].cl];
    const long rows = (long)NPIECE * n1;
    knn3_kernel<<<(int)((rows + 255) / 256), 256, 0, stream>>>(
        XYZ[spec[i].fl], XYZ[spec[i].cl], n1, n2, idx3, w3, rows);
    const int Kc = spec[i].C1 + spec[i].C2;
    float* hcat = (float*)bufB;                  // bufB is dead here; reuse as row-major
    const long tot = rows * Kc;
    interp_concat_kernel<<<(int)((tot + 255) / 256), 256, 0, stream>>>(
        spec[i].f1, spec[i].C1, spec[i].f2, spec[i].C2, idx3, w3, rows, n1, n2, hcat);
    const int Kpad = (Kc + 31) & ~31;
    const int Kt = Kpad / 32;
    const int Mtiles = (int)(rows / 16);
    pack_rows_kernel<<<(Mtiles * 32 + 255) / 256, 256, 0, stream>>>(
        hcat, rows, Kc, Kt, Mtiles, bufA);
    run_mlp(stream, fpW[i], fpB[i], fpDims[i], fpNL[i], Kc, (int)rows,
            bufA, bufB, Wp, spec[i].out, nullptr, 1, 1, 0);
  }

  // ------------------------- head -------------------------
  head_kernel<<<(NPIECE * 4096 + 255) / 256, 256, 0, stream>>>(
      FN0, headW, headB, (float*)d_out, (long)NPIECE * 4096);
}
